// MixedPrecisionQATLinearEnhanced_29669634081253
// MI455X (gfx1250) — matrix operand order repaired and checked by reading, not compiled, not run on hardware
//
#include <hip/hip_runtime.h>
#include <stdint.h>

#define B_    4
#define S_    2048
#define M_TOT (B_ * S_)   // 8192
#define K_TOT 4096
#define N_TOT 4096
#define EPS_F 1e-6f

#define BM 128
#define BN 128
#define BK 64

typedef __attribute__((ext_vector_type(16))) __bf16 v16bf;
typedef __attribute__((ext_vector_type(8)))  __bf16 v8bf;
typedef __attribute__((ext_vector_type(8)))  float  v8f;

// ---------------- helpers ----------------

__device__ __forceinline__ unsigned short f2bf_rne(float f) {
  unsigned int u = __float_as_uint(f);
  u += 0x7FFFu + ((u >> 16) & 1u);     // round-to-nearest-even to bf16
  return (unsigned short)(u >> 16);
}

// ---------------- pass 1: scalar params ----------------
// P[0..2]=aa softmax  P[3..5]=aw softmax  P[6]=scale P[7]=1/scale
// P[8]=s_a P[9]=beta P[10]=alpha P[11]=alpha/255 P[12]=s_w P[13]=amax(|W|) accum (uint bits)
__global__ void scalars_kernel(const float* __restrict__ lw, const float* __restrict__ la,
                               const float* __restrict__ rs, const float* __restrict__ wsc,
                               const float* __restrict__ asc, const float* __restrict__ ab,
                               const float* __restrict__ pa, float* __restrict__ P) {
  if (threadIdx.x != 0 || blockIdx.x != 0) return;
  const float ti = 1.0f / (5.0f * 0.7f);  // 1/(max(TEMP,1e-6)*0.7)
  {
    float a0 = la[0] * ti, a1 = la[1] * ti, a2 = la[2] * ti;
    float mx = fmaxf(a0, fmaxf(a1, a2));
    float e0 = expf(a0 - mx), e1 = expf(a1 - mx), e2 = expf(a2 - mx);
    float inv = 1.0f / (e0 + e1 + e2);
    P[0] = e0 * inv; P[1] = e1 * inv; P[2] = e2 * inv;
  }
  {
    float a0 = lw[0] * ti, a1 = lw[1] * ti, a2 = lw[2] * ti;
    float mx = fmaxf(a0, fmaxf(a1, a2));
    float e0 = expf(a0 - mx), e1 = expf(a1 - mx), e2 = expf(a2 - mx);
    float inv = 1.0f / (e0 + e1 + e2);
    P[3] = e0 * inv; P[4] = e1 * inv; P[5] = e2 * inv;
  }
  float sc = fmaxf(rs[0], EPS_F);
  P[6] = sc; P[7] = 1.0f / sc;
  P[8] = fmaxf(asc[0], EPS_F);
  P[9] = ab[0];
  float al = fmaxf(pa[0], EPS_F);
  P[10] = al; P[11] = al / 255.0f;
  P[12] = fmaxf(wsc[0], EPS_F);
  P[13] = 0.0f;                          // amax accumulator (bits == 0)
}

// ---------------- pass 2: amax(|W|) ----------------
__global__ __launch_bounds__(256)
void wamax_kernel(const float* __restrict__ W, float* __restrict__ P, int n4) {
  __shared__ float red[256];
  float m = 0.0f;
  for (size_t i = (size_t)blockIdx.x * 256 + threadIdx.x; i < (size_t)n4;
       i += (size_t)gridDim.x * 256) {
    float4 v = *(const float4*)(W + i * 4);
    m = fmaxf(m, fmaxf(fmaxf(fabsf(v.x), fabsf(v.y)),
                       fmaxf(fabsf(v.z), fabsf(v.w))));
  }
  red[threadIdx.x] = m;
  __syncthreads();
  for (int s = 128; s > 0; s >>= 1) {
    if (threadIdx.x < s) red[threadIdx.x] = fmaxf(red[threadIdx.x], red[threadIdx.x + s]);
    __syncthreads();
  }
  if (threadIdx.x == 0)
    atomicMax((unsigned int*)(P + 13), __float_as_uint(red[0]));
}

// ---------------- pass 3: quantize W -> bf16 ----------------
__global__ __launch_bounds__(256)
void quant_w_kernel(const float* __restrict__ W, const float* __restrict__ P,
                    unsigned short* __restrict__ qW) {
  size_t i = ((size_t)blockIdx.x * 256 + threadIdx.x) * 4;
  float aw0 = P[3], aw1 = P[4], aw2 = P[5];
  float inv_scale = P[7], s_w = P[12];
  float amax = __uint_as_float(((const unsigned int*)P)[13]) * inv_scale;
  float s8 = fmaxf(amax / 127.0f, EPS_F);
  float4 v = *(const float4*)(W + i);
  float in[4] = {v.x, v.y, v.z, v.w};
  unsigned long long pack = 0;
#pragma unroll
  for (int j = 0; j < 4; ++j) {
    float w = in[j] * inv_scale;
    float l = rintf(fminf(fmaxf(w / s_w, -8.0f), 7.0f)) * s_w;          // lsq 4b
    float u = fminf(fmaxf(rintf(w / s8), -128.0f), 127.0f) * s8;        // uniform-sym 8b
    float q = aw0 * l + aw1 * u + aw2 * w;
    pack |= ((unsigned long long)f2bf_rne(q)) << (16 * j);
  }
  *(unsigned long long*)(qW + i) = pack;
}

// ---------------- pass 4: quantize x -> bf16 ----------------
__global__ __launch_bounds__(256)
void quant_x_kernel(const float* __restrict__ x, const float* __restrict__ P,
                    unsigned short* __restrict__ qX) {
  size_t i = ((size_t)blockIdx.x * 256 + threadIdx.x) * 4;
  float aa0 = P[0], aa1 = P[1], aa2 = P[2];
  float scale = P[6], s_a = P[8], beta = P[9], alpha = P[10], step = P[11];
  float4 v = *(const float4*)(x + i);
  float in[4] = {v.x, v.y, v.z, v.w};
  unsigned long long pack = 0;
#pragma unroll
  for (int j = 0; j < 4; ++j) {
    float xf = in[j] * scale;
    float l = rintf(fminf(fmaxf((xf - beta) / s_a, -8.0f), 7.0f)) * s_a + beta; // lsq 4b
    float y = fminf(fmaxf(xf, 0.0f), alpha);
    float p = rintf(y / step) * step;                                           // pact 8b
    float q = aa0 * l + aa1 * p + aa2 * xf;
    pack |= ((unsigned long long)f2bf_rne(q)) << (16 * j);
  }
  *(unsigned long long*)(qX + i) = pack;
}

// ---------------- pass 5: bf16 WMMA GEMM  out = qX @ qW^T + b ----------------
// qX: [M,K] bf16 (K contiguous), qW: [N,K] bf16 (K contiguous)  -> NT GEMM
__global__ __launch_bounds__(256)
void qgemm_wmma(const unsigned short* __restrict__ qX,
                const unsigned short* __restrict__ qW,
                const float* __restrict__ bias,
                float* __restrict__ out) {
  __shared__ __attribute__((aligned(128))) unsigned short As[2][BM * BK];
  __shared__ __attribute__((aligned(128))) unsigned short Bs[2][BN * BK];

  const int tid  = threadIdx.x;
  const int lane = tid & 31;
  const int wid  = tid >> 5;     // 8 waves
  const int waveM = wid >> 1;    // 0..3  (32 rows each)
  const int waveN = wid & 1;     // 0..1  (64 cols each)
  const int r = lane & 15;
  const int h = lane >> 4;
  const int m0 = blockIdx.y * BM;
  const int n0 = blockIdx.x * BN;

  const unsigned short* gA = qX + (size_t)m0 * K_TOT;
  const unsigned short* gB = qW + (size_t)n0 * K_TOT;

  v8f acc[2][4];
#pragma unroll
  for (int mt = 0; mt < 2; ++mt)
#pragma unroll
    for (int nt = 0; nt < 4; ++nt)
#pragma unroll
      for (int e = 0; e < 8; ++e) acc[mt][nt][e] = 0.0f;

  // async global -> LDS tile fetch (CDNA5 async path, tracked by ASYNCcnt)
  auto issue = [&](const unsigned short* a, const unsigned short* b, int buf) {
    uint32_t la0 = (uint32_t)(uintptr_t)&As[buf][0];
    uint32_t lb0 = (uint32_t)(uintptr_t)&Bs[buf][0];
#pragma unroll
    for (int c = 0; c < 4; ++c) {
      int id  = c * 256 + tid;           // 1024 chunks of 16B per tile
      int row = id >> 3;                 // 8 chunks per 128B row (BK=64 bf16)
      int col = (id & 7) * 8;
      uint32_t loff = (uint32_t)(row * BK + col) * 2u;
      uint64_t ga = (uint64_t)(uintptr_t)(a + (size_t)row * K_TOT + col);
      uint64_t gb = (uint64_t)(uintptr_t)(b + (size_t)row * K_TOT + col);
      asm volatile("global_load_async_to_lds_b128 %0, %1, off"
                   :: "v"(la0 + loff), "v"(ga) : "memory");
      asm volatile("global_load_async_to_lds_b128 %0, %1, off"
                   :: "v"(lb0 + loff), "v"(gb) : "memory");
    }
  };

  issue(gA, gB, 0);
  const int nKT = K_TOT / BK;            // 64
  for (int kt = 0; kt < nKT; ++kt) {
    asm volatile("s_wait_asynccnt 0x0" ::: "memory");
    __syncthreads();
    const int buf = kt & 1;
    if (kt + 1 < nKT)
      issue(gA + (size_t)(kt + 1) * BK, gB + (size_t)(kt + 1) * BK, buf ^ 1);

    const unsigned short* Ab = &As[buf][0];
    const unsigned short* Bb = &Bs[buf][0];
#pragma unroll
    for (int kk = 0; kk < BK; kk += 32) {
      // A frag (16x32 bf16): lanes 0-15 hold K 0..15, lanes 16-31 K 16..31
      v16bf afr[2];
#pragma unroll
      for (int mt = 0; mt < 2; ++mt) {
        const unsigned short* p = Ab + (waveM * 32 + mt * 16 + r) * BK + kk + h * 16;
        v8bf lo = *(const v8bf*)p;
        v8bf hi = *(const v8bf*)(p + 8);
        afr[mt] = __builtin_shufflevector(lo, hi, 0, 1, 2, 3, 4, 5, 6, 7,
                                          8, 9, 10, 11, 12, 13, 14, 15);
      }
      // B frag (32x16 bf16): lane n holds K kh*16 .. kh*16+15 of column n (contiguous 32B)
      v16bf bfr[4];
#pragma unroll
      for (int nt = 0; nt < 4; ++nt) {
        const unsigned short* p = Bb + (waveN * 64 + nt * 16 + r) * BK + kk + h * 16;
        bfr[nt] = *(const v16bf*)p;
      }
#pragma unroll
      for (int mt = 0; mt < 2; ++mt)
#pragma unroll
        for (int nt = 0; nt < 4; ++nt)
          acc[mt][nt] = __builtin_amdgcn_wmma_f32_16x16x32_bf16(
              false, afr[mt], false, bfr[nt], (short)0, acc[mt][nt], false, false);
    }
  }

  // epilogue: C layout — VGPR e holds M = e + 8*(lane/16), N = lane&15
#pragma unroll
  for (int nt = 0; nt < 4; ++nt) {
    int n = n0 + waveN * 64 + nt * 16 + r;
    float bv = bias[n];
#pragma unroll
    for (int mt = 0; mt < 2; ++mt) {
      int mbase = m0 + waveM * 32 + mt * 16 + h * 8;
#pragma unroll
      for (int e = 0; e < 8; ++e)
        out[(size_t)(mbase + e) * N_TOT + n] = acc[mt][nt][e] + bv;
    }
  }
}

// ---------------- host launcher ----------------
extern "C" void kernel_launch(void* const* d_in, const int* in_sizes, int n_in,
                              void* d_out, int out_size, void* d_ws, size_t ws_size,
                              hipStream_t stream) {
  const float* x   = (const float*)d_in[0];
  const float* W   = (const float*)d_in[1];
  const float* b   = (const float*)d_in[2];
  const float* lw  = (const float*)d_in[3];
  const float* la  = (const float*)d_in[4];
  const float* rs  = (const float*)d_in[5];
  const float* wsc = (const float*)d_in[6];
  const float* asc = (const float*)d_in[7];
  const float* ab  = (const float*)d_in[8];
  const float* pa  = (const float*)d_in[9];
  float* out = (float*)d_out;

  float* P = (float*)d_ws;
  unsigned short* qX = (unsigned short*)((char*)d_ws + 256);
  unsigned short* qW = (unsigned short*)((char*)d_ws + 256 + (size_t)M_TOT * K_TOT * 2);

  scalars_kernel<<<1, 32, 0, stream>>>(lw, la, rs, wsc, asc, ab, pa, P);
  wamax_kernel<<<1024, 256, 0, stream>>>(W, P, (int)((size_t)N_TOT * K_TOT / 4));
  quant_w_kernel<<<(int)(((size_t)N_TOT * K_TOT) / 1024), 256, 0, stream>>>(W, P, qW);
  quant_x_kernel<<<(int)(((size_t)M_TOT * K_TOT) / 1024), 256, 0, stream>>>(x, P, qX);

  dim3 grid(N_TOT / BN, M_TOT / BM);   // 32 x 64
  qgemm_wmma<<<grid, 256, 0, stream>>>(qX, qW, b, out);
}
